// GDROS_7584912245500
// MI455X (gfx1250) — matrix operand order repaired and checked by reading, not compiled, not run on hardware
//
#include <hip/hip_runtime.h>
#include <cstdint>
#include <cstddef>

#define P   3072
#define HH  48
#define WW  64
#define CCH 256
#define REC 4080   // per-pixel pyramid record length: 3072+768+192+48
#define KC  512    // K-chunk staged in LDS per TDM transfer

typedef _Float16 v16h __attribute__((ext_vector_type(16)));
typedef _Float16 v8h  __attribute__((ext_vector_type(8)));
typedef float    v8f  __attribute__((ext_vector_type(8)));
typedef unsigned int u32x4 __attribute__((ext_vector_type(4)));
typedef int          i32x4 __attribute__((ext_vector_type(4)));
typedef int          i32x8 __attribute__((ext_vector_type(8)));

// ---------------------------------------------------------------------------
// 16-bit A-matrix (16x32) fragment: two 16B loads per lane per K-step
// ---------------------------------------------------------------------------
__device__ __forceinline__ v16h load_a16(const _Float16* base, int kb) {
  v8h lo = *(const v8h*)(base + kb);
  v8h hi = *(const v8h*)(base + kb + 16);
  v16h r;
#pragma unroll
  for (int j = 0; j < 8; ++j) { r[j] = lo[j]; r[j + 8] = hi[j]; }
  return r;
}

__device__ __forceinline__ v8f wmma_f16(v16h a, v16h b, v8f c) {
  return __builtin_amdgcn_wmma_f32_16x16x32_f16(false, a, false, b,
                                                (short)0, c, false, false);
}

// ---------------------------------------------------------------------------
// WMMA GEMM with TDM-staged B panel.
//   A  [M][Kpad] f16 (global, streamed)
//   Bw [N][Kpad] f16 (global; 32-row panel DMA'd to LDS per workgroup)
// Workgroup: 128 threads = 4 waves; wave w computes a 32x32 output block
// (2x2 tiles of 16x16) at rows (blockIdx.y*4+w)*32, cols blockIdx.x*32.
// out = act( scale * (A x Bw^T + bias) )
// outT==0 : out[m*ldo + n]   (correlation volume layout)
// outT==1 : out[n*ldo + m]   (conv channel-major layout, ldo = P)
// ---------------------------------------------------------------------------
__global__ void __launch_bounds__(128)
k_gemm16(const _Float16* __restrict__ A,
         const _Float16* __restrict__ Bw,
         const float* __restrict__ bias, int Ob,
         float* __restrict__ out, int ldo, int Kpad, int Ntiles,
         int act, float scale, int outT)
{
  __shared__ _Float16 Bs[32 * KC];

  const int lane = threadIdx.x & 31;
  const int wv   = threadIdx.x >> 5;
  const int ln   = lane & 15;
  const int hf   = lane >> 4;

  const int tileN0 = blockIdx.x * 2;
  const bool hasN1 = (tileN0 + 1) < Ntiles;
  const int tileM0 = (blockIdx.y * 4 + wv) * 2;
  const int row0   = blockIdx.x * 32;                 // first B row staged

  const unsigned ldsBase = (unsigned)(uintptr_t)&Bs[0];

  const _Float16* arow0 = A + (size_t)(tileM0 * 16 + ln) * Kpad;
  const _Float16* arow1 = arow0 + (size_t)16 * Kpad;

  v8f acc00 = {0.f,0.f,0.f,0.f,0.f,0.f,0.f,0.f};
  v8f acc01 = acc00, acc10 = acc00, acc11 = acc00;

  for (int kbase = 0; kbase < Kpad; kbase += KC) {
    const int rem = Kpad - kbase;
    const int Kc  = (rem < KC) ? rem : KC;

    if (wv == 0) {
      // --- Tensor Data Mover: 2D tile (Kc x 32 rows) of Bw -> LDS ---------
      unsigned long long ga =
          (unsigned long long)(uintptr_t)Bw + ((size_t)row0 * Kpad + kbase) * 2ull;
      unsigned td0 = (unsigned)rem;                 // dim0 extent (OOB guard)
      unsigned rowsLeft = (unsigned)(Ntiles * 16 - row0);
      unsigned td1 = rowsLeft;                      // rows beyond Opad -> zero

      u32x4 g0 = {0u, 0u, 0u, 0u};
      g0[0] = 1u;                                    // count=1, user descriptor
      g0[1] = ldsBase;                               // lds_addr (bytes)
      g0[2] = (unsigned)(ga & 0xFFFFFFFFu);          // global_addr[31:0]
      g0[3] = (unsigned)((ga >> 32) & 0x1FFFFFFu)    // global_addr[56:32]
              | (2u << 30);                          // type = 2 ("image")
      i32x8 g1 = {0,0,0,0,0,0,0,0};
      g1[0] = (int)(1u << 16);                       // data_size = 1 -> 2 bytes
      g1[1] = (int)((td0 & 0xFFFFu) << 16);          // tensor_dim0[15:0]
      g1[2] = (int)(((td0 >> 16) & 0xFFFFu) |        // tensor_dim0[31:16]
                    ((td1 & 0xFFFFu) << 16));        // tensor_dim1[15:0]
      g1[3] = (int)(((td1 >> 16) & 0xFFFFu) |        // tensor_dim1[31:16]
                    ((unsigned)Kc << 16));           // tile_dim0 = Kc
      g1[4] = 32;                                    // tile_dim1=32, tile_dim2=0
      g1[5] = (int)(unsigned)Kpad;                   // tensor_dim0_stride[31:0]
      g1[6] = 0;                                     // stride hi / dim1_stride lo
      g1[7] = 0;
      i32x4 zz = {0, 0, 0, 0};
#if defined(__clang_major__) && (__clang_major__ >= 23)
      i32x8 z8 = {0,0,0,0,0,0,0,0};
      __builtin_amdgcn_tensor_load_to_lds(g0, g1, zz, zz, z8, 0);
#else
      __builtin_amdgcn_tensor_load_to_lds(g0, g1, zz, zz, 0);
#endif
      __builtin_amdgcn_s_wait_tensorcnt(0);
    }
    __syncthreads();

    // LDS panel layout: [32 rows][Kc], row-contiguous (TDM packing order)
    const _Float16* b0p = Bs + (size_t)ln * Kc;
    const _Float16* b1p = Bs + (size_t)(16 + ln) * Kc;

    for (int kc = 0; kc < Kc; kc += 32) {
      v16h a0 = load_a16(arow0, kbase + kc + hf * 8);
      v16h a1 = load_a16(arow1, kbase + kc + hf * 8);
      v16h b0 = load_a16(b0p, kc + hf * 8);
      acc00 = wmma_f16(a0, b0, acc00);
      acc10 = wmma_f16(a1, b0, acc10);
      if (hasN1) {
        v16h b1 = load_a16(b1p, kc + hf * 8);
        acc01 = wmma_f16(a0, b1, acc01);
        acc11 = wmma_f16(a1, b1, acc11);
      }
    }
    __syncthreads();
  }

  // ---- epilogue: bias + activation + store ------------------------------
  const int n0 = tileN0 * 16 + ln;
  const int n1 = n0 + 16;
  const float bv0 = (bias != nullptr && n0 < Ob) ? bias[n0] : 0.f;
  const float bv1 = (bias != nullptr && n1 < Ob) ? bias[n1] : 0.f;

  auto emit = [&](v8f acc, int mt, int n, float bv) {
#pragma unroll
    for (int r = 0; r < 8; ++r) {
      int m   = mt * 16 + hf * 8 + r;
      float v = scale * (acc[r] + bv);
      if      (act == 1) v = fmaxf(v, 0.f);
      else if (act == 2) v = 1.f / (1.f + __expf(-v));
      else if (act == 3) v = tanhf(v);
      if (outT) out[(size_t)n * ldo + m] = v;
      else      out[(size_t)m * ldo + n] = v;
    }
  };
  emit(acc00, tileM0,     n0, bv0);
  emit(acc10, tileM0 + 1, n0, bv0);
  if (hasN1) {
    emit(acc01, tileM0,     n1, bv1);
    emit(acc11, tileM0 + 1, n1, bv1);
  }
}

// ---------------------------------------------------------------------------
// fmap (C,P) f32  ->  (P,C) f16  (GEMM operand layout)
// ---------------------------------------------------------------------------
__global__ void k_f16t(const float* __restrict__ src, _Float16* __restrict__ dst, int n) {
  int i = blockIdx.x * blockDim.x + threadIdx.x;
  if (i >= n) return;
  int c = i / P, p = i - c * P;
  dst[(size_t)p * CCH + c] = (_Float16)src[i];
}

// weights f32 (O,K) -> f16 (Opad,Kpad), zero padded
__global__ void k_wconv(const float* __restrict__ w, _Float16* __restrict__ dst,
                        int O, int K, int Opad, int Kpad) {
  int i = blockIdx.x * blockDim.x + threadIdx.x;
  int n = Opad * Kpad;
  if (i >= n) return;
  int o = i / Kpad, k = i - o * Kpad;
  float v = (o < O && k < K) ? w[(size_t)o * K + k] : 0.f;
  dst[i] = (_Float16)v;
}

// im2col: src [Cin][48][64] f32 -> A [P][Kpad] f16, SAME padding (KH/2, KW/2)
__global__ void k_im2col(const float* __restrict__ src, _Float16* __restrict__ A,
                         int Cin, int KH, int KW, int Kpad) {
  int i = blockIdx.x * blockDim.x + threadIdx.x;
  int n = P * Kpad;
  if (i >= n) return;
  int p = i / Kpad, k = i - p * Kpad;
  int K = Cin * KH * KW;
  float v = 0.f;
  if (k < K) {
    int ci  = k / (KH * KW);
    int rkk = k - ci * (KH * KW);
    int kh  = rkk / KW;
    int kw  = rkk - kh * KW;
    int y = p / WW + kh - KH / 2;
    int x = p % WW + kw - KW / 2;
    if (y >= 0 && y < HH && x >= 0 && x < WW)
      v = src[(size_t)ci * P + y * WW + x];
  }
  A[i] = (_Float16)v;
}

// 2x2 average pooling inside each pixel's pyramid record
__global__ void k_pool(float* __restrict__ corr, int srcOff, int dstOff,
                       int Hs, int Ws) {
  int hd = Hs >> 1, wd = Ws >> 1;
  int n = P * hd * wd;
  int i = blockIdx.x * blockDim.x + threadIdx.x;
  if (i >= n) return;
  int p = i / (hd * wd), r = i - p * (hd * wd);
  int y = r / wd, x = r - y * wd;
  const float* s = corr + (size_t)p * REC + srcOff;
  float v = 0.25f * (s[(2 * y) * Ws + 2 * x]     + s[(2 * y) * Ws + 2 * x + 1] +
                     s[(2 * y + 1) * Ws + 2 * x] + s[(2 * y + 1) * Ws + 2 * x + 1]);
  corr[(size_t)p * REC + dstOff + y * wd + x] = v;
}

// net=tanh(cnet[:128]); ctx=relu(cnet[128:]) written into hx/rx ctx slots
__global__ void k_init_netctx(const float* __restrict__ cnet,
                              float* __restrict__ net,
                              float* __restrict__ hx,
                              float* __restrict__ rx) {
  int i = blockIdx.x * blockDim.x + threadIdx.x;
  if (i >= 128 * P) return;
  net[i] = tanhf(cnet[i]);
  float c = fmaxf(cnet[128 * P + i], 0.f);
  hx[128 * P + i] = c;
  rx[128 * P + i] = c;
}

__global__ void k_init_coords(float* __restrict__ coords) {
  int i = blockIdx.x * blockDim.x + threadIdx.x;
  if (i >= P) return;
  coords[i]     = (float)(i % WW);
  coords[P + i] = (float)(i / WW);
}

// flow = coords1 - coords0
__global__ void k_flow(const float* __restrict__ coords, float* __restrict__ flow) {
  int i = blockIdx.x * blockDim.x + threadIdx.x;
  if (i >= 2 * P) return;
  int ch = i / P, p = i - ch * P;
  float base = ch ? (float)(p / WW) : (float)(p % WW);
  flow[i] = coords[i] - base;
}

// correlation lookup: bilinear sample 81 offsets x 4 levels per pixel
__global__ void k_corr_lookup(const float* __restrict__ corr,
                              const float* __restrict__ coords,
                              float* __restrict__ cf) {
  int i = blockIdx.x * blockDim.x + threadIdx.x;
  if (i >= P * 81) return;
  int p = i / 81, o = i - p * 81;
  int dyi = o / 9, dxi = o - dyi * 9;
  float dx = (float)(dxi - 4), dy = (float)(dyi - 4);
  float cx = coords[p], cy = coords[P + p];
  const int lvH[4] = {48, 24, 12, 6};
  const int lvW[4] = {64, 32, 16, 8};
  const int lvO[4] = {0, 3072, 3840, 4032};
#pragma unroll
  for (int l = 0; l < 4; ++l) {
    float inv = 1.f / (float)(1 << l);
    float X = cx * inv + dx, Y = cy * inv + dy;
    int Hl = lvH[l], Wl = lvW[l];
    const float* img = corr + (size_t)p * REC + lvO[l];
    float x0f = floorf(X), y0f = floorf(Y);
    float wx1 = X - x0f, wx0 = 1.f - wx1;
    float wy1 = Y - y0f, wy0 = 1.f - wy1;
    float val = 0.f;
#pragma unroll
    for (int cy2 = 0; cy2 < 2; ++cy2) {
#pragma unroll
      for (int cx2 = 0; cx2 < 2; ++cx2) {
        float xx = x0f + (float)cx2, yy = y0f + (float)cy2;
        float g = 0.f;
        if (xx >= 0.f && xx <= (float)(Wl - 1) && yy >= 0.f && yy <= (float)(Hl - 1)) {
          int xi = (int)fminf(fmaxf(xx, 0.f), (float)(Wl - 1));
          int yi = (int)fminf(fmaxf(yy, 0.f), (float)(Hl - 1));
          g = img[yi * Wl + xi];
        }
        val += (cx2 ? wx1 : wx0) * (cy2 ? wy1 : wy0) * g;
      }
    }
    cf[(size_t)(l * 81 + o) * P + p] = val;
  }
}

__global__ void k_copy(float* __restrict__ d, const float* __restrict__ s, int n) {
  int i = blockIdx.x * blockDim.x + threadIdx.x;
  if (i < n) d[i] = s[i];
}
__global__ void k_mul(float* __restrict__ d, const float* __restrict__ a,
                      const float* __restrict__ b, int n) {
  int i = blockIdx.x * blockDim.x + threadIdx.x;
  if (i < n) d[i] = a[i] * b[i];
}
__global__ void k_gru(float* __restrict__ net, const float* __restrict__ z,
                      const float* __restrict__ q, int n) {
  int i = blockIdx.x * blockDim.x + threadIdx.x;
  if (i < n) net[i] = (1.f - z[i]) * net[i] + z[i] * q[i];
}
__global__ void k_axpy(float* __restrict__ d, const float* __restrict__ s, int n) {
  int i = blockIdx.x * blockDim.x + threadIdx.x;
  if (i < n) d[i] += s[i];
}

// convex upsampling: out [2][384][512]
__global__ void k_upsample(const float* __restrict__ flow,
                           const float* __restrict__ mask,
                           float* __restrict__ out) {
  int gid = blockIdx.x * blockDim.x + threadIdx.x;
  if (gid >= P * 64) return;
  int p = gid / 64, q = gid - p * 64;
  int ii = q / 8, jj = q - ii * 8;
  int y = p / WW, x = p - y * WW;
  float mv[9], mx = -1e30f;
#pragma unroll
  for (int k = 0; k < 9; ++k) {
    mv[k] = mask[(size_t)(k * 64 + ii * 8 + jj) * P + p];
    mx = fmaxf(mx, mv[k]);
  }
  float s = 0.f;
#pragma unroll
  for (int k = 0; k < 9; ++k) { mv[k] = __expf(mv[k] - mx); s += mv[k]; }
  float f0 = 0.f, f1 = 0.f;
#pragma unroll
  for (int k = 0; k < 9; ++k) {
    int di = k / 3, dj = k - di * 3;
    int yy = y + di - 1, xx = x + dj - 1;
    if (yy >= 0 && yy < HH && xx >= 0 && xx < WW) {
      float w = mv[k];
      f0 += w * 8.f * flow[yy * WW + xx];
      f1 += w * 8.f * flow[P + yy * WW + xx];
    }
  }
  f0 /= s; f1 /= s;
  int oy = y * 8 + ii, ox = x * 8 + jj;
  out[(size_t)oy * 512 + ox]               = f0;
  out[(size_t)384 * 512 + oy * 512 + ox]   = f1;
}

// ---------------------------------------------------------------------------
// Host orchestration
// ---------------------------------------------------------------------------
struct CW { int wi, bi, O, Cin, KH, KW; int Opad, K, Kpad; size_t woff; };

static inline dim3 gr1(int n) { return dim3((unsigned)((n + 255) / 256)); }

extern "C" void kernel_launch(void* const* d_in, const int* in_sizes, int n_in,
                              void* d_out, int out_size, void* d_ws, size_t ws_size,
                              hipStream_t stream) {
  (void)in_sizes; (void)n_in; (void)out_size; (void)ws_size;
  char* ws = (char*)d_ws;
  const int ITERS = 12;

  // ---- workspace layout -------------------------------------------------
  size_t off = 0;
  auto bump = [&](size_t bytes) {
    off = (off + 255) & ~(size_t)255;
    size_t o = off; off += bytes; return o;
  };
  size_t o_fm1t = bump((size_t)P * CCH * 2);
  size_t o_fm2t = bump((size_t)P * CCH * 2);
  size_t o_corr = bump((size_t)P * REC * 4);
  size_t o_A    = bump((size_t)P * 2304 * 2);

  CW cw[15] = {
    {3, 4, 256, 324, 1, 1}, {5, 6, 192, 256, 3, 3}, {7, 8, 128, 2, 7, 7},
    {9, 10, 64, 128, 3, 3}, {11, 12, 126, 256, 3, 3},
    {13, 14, 128, 384, 1, 5}, {15, 16, 128, 384, 1, 5}, {17, 18, 128, 384, 1, 5},
    {19, 20, 128, 384, 5, 1}, {21, 22, 128, 384, 5, 1}, {23, 24, 128, 384, 5, 1},
    {25, 26, 256, 128, 3, 3}, {27, 28, 2, 256, 3, 3},
    {29, 30, 256, 128, 3, 3}, {31, 32, 576, 256, 1, 1},
  };
  for (int i = 0; i < 15; ++i) {
    cw[i].K    = cw[i].Cin * cw[i].KH * cw[i].KW;
    cw[i].Kpad = (cw[i].K + 31) & ~31;
    cw[i].Opad = (cw[i].O + 15) & ~15;
    cw[i].woff = bump((size_t)cw[i].Opad * cw[i].Kpad * 2);
  }

  size_t o_cf     = bump((size_t)324 * P * 4);
  size_t o_flow   = bump((size_t)2 * P * 4);
  size_t o_coords = bump((size_t)2 * P * 4);
  size_t o_cor1   = bump((size_t)256 * P * 4);
  size_t o_catcf  = bump((size_t)256 * P * 4);
  size_t o_flo1   = bump((size_t)128 * P * 4);
  size_t o_hx     = bump((size_t)384 * P * 4);   // [net | ctx | motfl]
  size_t o_rx     = bump((size_t)384 * P * 4);   // [r*net | ctx | motfl]
  size_t o_net    = bump((size_t)128 * P * 4);
  size_t o_z      = bump((size_t)128 * P * 4);
  size_t o_r      = bump((size_t)128 * P * 4);
  size_t o_q      = bump((size_t)128 * P * 4);
  size_t o_fh1o   = bump((size_t)256 * P * 4);
  size_t o_delta  = bump((size_t)16 * P * 4);
  size_t o_m1o    = bump((size_t)256 * P * 4);
  size_t o_mask   = bump((size_t)576 * P * 4);

  auto F  = [&](size_t o) { return (float*)(ws + o); };
  auto Hp = [&](size_t o) { return (_Float16*)(ws + o); };

  // ---- setup: f16 operands ---------------------------------------------
  k_f16t<<<gr1(CCH * P), 256, 0, stream>>>((const float*)d_in[0], Hp(o_fm1t), CCH * P);
  k_f16t<<<gr1(CCH * P), 256, 0, stream>>>((const float*)d_in[1], Hp(o_fm2t), CCH * P);
  for (int i = 0; i < 15; ++i) {
    int n = cw[i].Opad * cw[i].Kpad;
    k_wconv<<<gr1(n), 256, 0, stream>>>((const float*)d_in[cw[i].wi], Hp(cw[i].woff),
                                        cw[i].O, cw[i].K, cw[i].Opad, cw[i].Kpad);
  }

  // ---- correlation volume via WMMA, scaled by 1/sqrt(256) --------------
  {
    int Ntiles = P / 16;  // 192
    k_gemm16<<<dim3((Ntiles + 1) / 2, 24), 128, 0, stream>>>(
        Hp(o_fm1t), Hp(o_fm2t), nullptr, 0, F(o_corr), REC, CCH, Ntiles,
        /*act*/0, /*scale*/0.0625f, /*outT*/0);
  }

  // ---- pyramid ----------------------------------------------------------
  k_pool<<<gr1(P * 24 * 32), 256, 0, stream>>>(F(o_corr), 0,    3072, 48, 64);
  k_pool<<<gr1(P * 12 * 16), 256, 0, stream>>>(F(o_corr), 3072, 3840, 24, 32);
  k_pool<<<gr1(P * 6 * 8),   256, 0, stream>>>(F(o_corr), 3840, 4032, 12, 16);

  // ---- init net / ctx / coords -----------------------------------------
  k_init_netctx<<<gr1(128 * P), 256, 0, stream>>>((const float*)d_in[2], F(o_net),
                                                  F(o_hx), F(o_rx));
  k_init_coords<<<gr1(P), 256, 0, stream>>>(F(o_coords));

  // ---- conv helpers -----------------------------------------------------
  auto gemm_only = [&](int ci, float* dst, int act, float scale) {
    CW& c = cw[ci];
    int Ntiles = c.Opad / 16;
    k_gemm16<<<dim3((Ntiles + 1) / 2, 24), 128, 0, stream>>>(
        Hp(o_A), Hp(c.woff), (const float*)d_in[c.bi], c.O,
        dst, P, c.Kpad, Ntiles, act, scale, 1);
  };
  auto conv = [&](int ci, const float* src, float* dst, int act, float scale) {
    CW& c = cw[ci];
    k_im2col<<<gr1(P * c.Kpad), 256, 0, stream>>>(src, Hp(o_A), c.Cin, c.KH, c.KW, c.Kpad);
    gemm_only(ci, dst, act, scale);
  };

  // ---- iterations --------------------------------------------------------
  for (int it = 0; it < ITERS; ++it) {
    k_corr_lookup<<<gr1(P * 81), 256, 0, stream>>>(F(o_corr), F(o_coords), F(o_cf));
    k_flow<<<gr1(2 * P), 256, 0, stream>>>(F(o_coords), F(o_flow));

    conv(0, F(o_cf),   F(o_cor1),            1, 1.f);   // convc1 relu
    conv(1, F(o_cor1), F(o_catcf),           1, 1.f);   // convc2 relu (192ch)
    conv(2, F(o_flow), F(o_flo1),            1, 1.f);   // convf1 relu
    conv(3, F(o_flo1), F(o_catcf) + 192 * P, 1, 1.f);   // convf2 relu (64ch)
    conv(4, F(o_catcf), F(o_hx) + 256 * P,   1, 1.f);   // convm relu -> motfl
    // motfl tail = flow ; mirror motfl into rx
    k_copy<<<gr1(2 * P), 256, 0, stream>>>(F(o_hx) + (256 + 126) * P, F(o_flow), 2 * P);
    k_copy<<<gr1(128 * P), 256, 0, stream>>>(F(o_rx) + 256 * P, F(o_hx) + 256 * P, 128 * P);

    for (int g = 0; g < 2; ++g) {
      int zi = 5 + g * 3, ri = 6 + g * 3, qi = 7 + g * 3;
      k_copy<<<gr1(128 * P), 256, 0, stream>>>(F(o_hx), F(o_net), 128 * P);
      CW& cz = cw[zi];
      k_im2col<<<gr1(P * cz.Kpad), 256, 0, stream>>>(F(o_hx), Hp(o_A),
                                                     cz.Cin, cz.KH, cz.KW, cz.Kpad);
      gemm_only(zi, F(o_z), 2, 1.f);                    // z = sigmoid
      gemm_only(ri, F(o_r), 2, 1.f);                    // r = sigmoid
      k_mul<<<gr1(128 * P), 256, 0, stream>>>(F(o_rx), F(o_r), F(o_net), 128 * P);
      CW& cq = cw[qi];
      k_im2col<<<gr1(P * cq.Kpad), 256, 0, stream>>>(F(o_rx), Hp(o_A),
                                                     cq.Cin, cq.KH, cq.KW, cq.Kpad);
      gemm_only(qi, F(o_q), 3, 1.f);                    // q = tanh
      k_gru<<<gr1(128 * P), 256, 0, stream>>>(F(o_net), F(o_z), F(o_q), 128 * P);
    }

    // flow head + mask head share the im2col over net (both Cin=128, 3x3)
    CW& cf1 = cw[11];
    k_im2col<<<gr1(P * cf1.Kpad), 256, 0, stream>>>(F(o_net), Hp(o_A),
                                                    cf1.Cin, cf1.KH, cf1.KW, cf1.Kpad);
    gemm_only(11, F(o_fh1o), 1, 1.f);                   // fh1 relu
    gemm_only(13, F(o_m1o),  1, 1.f);                   // mask1 relu
    conv(12, F(o_fh1o), F(o_delta), 0, 1.f);            // fh2
    conv(14, F(o_m1o),  F(o_mask),  0, 0.25f);          // mask2 * 0.25

    k_axpy<<<gr1(2 * P), 256, 0, stream>>>(F(o_coords), F(o_delta), 2 * P);
    k_flow<<<gr1(2 * P), 256, 0, stream>>>(F(o_coords), F(o_flow));
    k_upsample<<<gr1(P * 64), 256, 0, stream>>>(F(o_flow), F(o_mask),
                                                (float*)d_out + (size_t)it * 2 * 384 * 512);
  }
}
